// Cross_Attention_17669495456332
// MI455X (gfx1250) — matrix operand order repaired and checked by reading, not compiled, not run on hardware
//
#include <hip/hip_runtime.h>
#include <hip/hip_bf16.h>

// ---------------------------------------------------------------------------
// Cross attention, B=16, N=1024, C=512.
//   S = dom @ ref^T * 0.125 ; P = softmax(S) ; x = P @ ref
//   scramble == view x^T ([C,N]) as [N,C] row-major
//   out = Xs @ W^T + b
// f16 WMMA (v_wmma_f32_16x16x32_f16), f32 accumulate.
// M=32 query tile per block to halve L2 traffic; dom tile staged to LDS with
// gfx1250 async loads (GLOBAL_LOAD_ASYNC_TO_LDS_B128) when available.
// Workspace (f16): dom16 | ref16 | refT16 | xsT16 | w16  (~68 MB)
// ---------------------------------------------------------------------------

#define BB 16
#define NN 1024
#define CC 512
#define SCALE 0.125f

typedef __attribute__((ext_vector_type(16))) _Float16 v16h;
typedef __attribute__((ext_vector_type(8)))  _Float16 v8h;
typedef __attribute__((ext_vector_type(8)))  float    v8f;
typedef __attribute__((ext_vector_type(4)))  int      v4i;

union HV { v16h v; v8h h[2]; };

static __device__ inline v8f wmma16(v16h a, v16h b, v8f c) {
  return __builtin_amdgcn_wmma_f32_16x16x32_f16(false, a, false, b,
                                                (short)0, c, false, false);
}

#if __has_builtin(__builtin_amdgcn_global_load_async_to_lds_b128) && \
    __has_builtin(__builtin_amdgcn_s_wait_asynccnt)
#define USE_ASYNC_LDS 1
#endif

// ---------------- stage 0: f32 -> f16 convert (elementwise) ----------------
__global__ void cvt_f16_kernel(const float* __restrict__ s,
                               _Float16* __restrict__ d, int n) {
  int stride = gridDim.x * blockDim.x;
  for (int i = blockIdx.x * blockDim.x + threadIdx.x; i < n; i += stride)
    d[i] = (_Float16)s[i];
}

// ------------- stage 0b: ref [B,N,C] f32 -> refT [B,C,N] f16 ---------------
__global__ void transpose_cvt_kernel(const float* __restrict__ src,
                                     _Float16* __restrict__ dst) {
  __shared__ _Float16 t[32][33];
  int b  = blockIdx.z;
  int m0 = blockIdx.x * 32;   // N tile
  int c0 = blockIdx.y * 32;   // C tile
  int tx = threadIdx.x, ty = threadIdx.y;   // 32 x 8
#pragma unroll
  for (int i = 0; i < 4; ++i) {
    int m = m0 + ty + i * 8;
    t[ty + i * 8][tx] = (_Float16)src[((size_t)b * NN + m) * CC + c0 + tx];
  }
  __syncthreads();
#pragma unroll
  for (int i = 0; i < 4; ++i) {
    int c = c0 + ty + i * 8;
    dst[((size_t)b * CC + c) * NN + m0 + tx] = t[tx][ty + i * 8];
  }
}

// ---------------- stage 1: fused attention for one 32-row tile -------------
// 512 threads = 16 waves.
// Phase 1: wave w -> m-strip ms=w>>3 (16 rows), S cols [ (w&7)*128, +128 )
// Phase 2: wave w -> O^T rows (C dim) [w*32, w*32+32), all 32 queries
__global__ __launch_bounds__(512)
void attn_kernel(const _Float16* __restrict__ dom16,
                 const _Float16* __restrict__ ref16,
                 const _Float16* __restrict__ refT16,
                 _Float16* __restrict__ xsT16) {
  __shared__ _Float16 As[32 * CC];       // staged dom tile, f16 (32 KB)
  __shared__ _Float16 Pl[32 * NN];       // unnormalized P, f16 (64 KB)
  __shared__ float redmax[16][16];
  __shared__ float redsum[16][16];
  __shared__ float rden[32];

  const int tid  = threadIdx.x;
  const int wid  = tid >> 5;
  const int lane = tid & 31;
  const int hi   = lane >> 4;     // half-wave select
  const int n    = lane & 15;     // column lane / A-row lane
  const int b    = blockIdx.x >> 5;
  const int m0   = (blockIdx.x & 31) * 32;   // query rows of this block
  const int ms   = wid >> 3;                 // phase-1 m-strip (0/1)
  const int cs   = wid & 7;                  // phase-1 col strip

  // ------- stage dom rows [m0, m0+32) x C into LDS (32 KB, contiguous) -----
  {
    const char* gsrc = (const char*)(dom16 + ((size_t)b * NN + m0) * CC);
#if defined(USE_ASYNC_LDS)
#pragma unroll
    for (int i = 0; i < 4; ++i) {
      int chunk = tid + i * 512;           // 2048 x 16B = 32 KB
      __builtin_amdgcn_global_load_async_to_lds_b128(
          (__attribute__((address_space(1))) v4i*)(gsrc + chunk * 16),
          (__attribute__((address_space(3))) v4i*)(((char*)As) + chunk * 16),
          0, 0);
    }
    __builtin_amdgcn_s_wait_asynccnt(0);
#else
#pragma unroll
    for (int i = 0; i < 4; ++i) {
      int chunk = tid + i * 512;
      ((v8h*)As)[chunk] = ((const v8h*)gsrc)[chunk];
    }
#endif
  }
  __syncthreads();

  // ---------------- phase 1: S = dom @ ref^T ----------------
  v8f acc[8];
#pragma unroll
  for (int t = 0; t < 8; ++t) acc[t] = v8f{0};

  const _Float16* Arow = &As[(ms * 16 + n) * CC];
  for (int kc = 0; kc < CC; kc += 32) {
    HV a;                                  // A: 16x32, lane n = row (from LDS)
    a.h[0] = *(const v8h*)(Arow + kc + 8 * hi);
    a.h[1] = *(const v8h*)(Arow + kc + 16 + 8 * hi);
#pragma unroll
    for (int t = 0; t < 8; ++t) {
      // B[k][n] = ref[col0+n][kc+k]; lane holds K = 8*hi+0..7, 16+8*hi+0..7
      const _Float16* bp =
          ref16 + ((size_t)b * NN + cs * 128 + t * 16 + n) * CC + kc;
      HV bv;
      bv.h[0] = *(const v8h*)(bp + 8 * hi);
      bv.h[1] = *(const v8h*)(bp + 16 + 8 * hi);
      acc[t] = wmma16(a.v, bv.v, acc[t]);
    }
  }

  // ---------------- softmax (strip-local rows r = 8*hi + j) ----------------
#pragma unroll
  for (int j = 0; j < 8; ++j) {
    float m = acc[0][j];
#pragma unroll
    for (int t = 1; t < 8; ++t) m = fmaxf(m, acc[t][j]);
#pragma unroll
    for (int s = 1; s < 16; s <<= 1) m = fmaxf(m, __shfl_xor(m, s, 32));
    if (n == j) redmax[wid][8 * hi + j] = m * SCALE;
  }
  __syncthreads();

#pragma unroll
  for (int j = 0; j < 8; ++j) {
    float m = -3.402823e38f;
#pragma unroll
    for (int w = 0; w < 8; ++w) m = fmaxf(m, redmax[ms * 8 + w][8 * hi + j]);
    float s = 0.f;
#pragma unroll
    for (int t = 0; t < 8; ++t) {
      float e = __expf(acc[t][j] * SCALE - m);
      acc[t][j] = e;
      s += e;
    }
#pragma unroll
    for (int sft = 1; sft < 16; sft <<= 1) s += __shfl_xor(s, sft, 32);
    if (n == j) redsum[wid][8 * hi + j] = s;
  }

  // stash unnormalized P into LDS as f16
#pragma unroll
  for (int t = 0; t < 8; ++t)
#pragma unroll
    for (int j = 0; j < 8; ++j)
      Pl[(ms * 16 + 8 * hi + j) * NN + cs * 128 + t * 16 + n] =
          (_Float16)acc[t][j];
  __syncthreads();

  if ((wid & 7) == 0) {
#pragma unroll
    for (int j = 0; j < 8; ++j)
      if (n == j) {
        float s = 0.f;
#pragma unroll
        for (int w = 0; w < 8; ++w) s += redsum[ms * 8 + w][8 * hi + j];
        rden[ms * 16 + 8 * hi + j] = 1.0f / s;
      }
  }
  __syncthreads();

  // ---------------- phase 2: O^T = V^T @ P^T ----------------
  v8f o[2][2] = {{v8f{0}, v8f{0}}, {v8f{0}, v8f{0}}};
  const int c0 = wid * 32;
  for (int kc = 0; kc < NN; kc += 32) {
    HV bq[2];
#pragma unroll
    for (int q = 0; q < 2; ++q) {
      // B[k][n] = P[q*16+n][kc+k]; lane holds K = 8*hi+0..7, 16+8*hi+0..7
      const _Float16* pb = &Pl[(q * 16 + n) * NN + kc];
      bq[q].h[0] = *(const v8h*)(pb + 8 * hi);
      bq[q].h[1] = *(const v8h*)(pb + 16 + 8 * hi);
    }
#pragma unroll
    for (int ct = 0; ct < 2; ++ct) {
      const _Float16* ap =
          refT16 + ((size_t)b * CC + c0 + ct * 16 + n) * NN + kc;
      HV a;
      a.h[0] = *(const v8h*)(ap + 8 * hi);
      a.h[1] = *(const v8h*)(ap + 16 + 8 * hi);
#pragma unroll
      for (int q = 0; q < 2; ++q)
        o[ct][q] = wmma16(a.v, bq[q].v, o[ct][q]);
    }
  }

#pragma unroll
  for (int q = 0; q < 2; ++q) {
    const float rn = rden[q * 16 + n];
    const int m = m0 + q * 16 + n;
#pragma unroll
    for (int ct = 0; ct < 2; ++ct)
#pragma unroll
      for (int j = 0; j < 8; ++j) {
        int c = c0 + ct * 16 + 8 * hi + j;
        xsT16[((size_t)b * CC + c) * NN + m] = (_Float16)(o[ct][q][j] * rn);
      }
  }
}

// ---------------- stage 2: projection GEMM + bias ----------------
// Xs = xsT16 viewed as [B*N, C] row-major. out = Xs @ W^T + b  (f32 out)
// Block: 256 thr = 8 waves; 32 output rows per block; wave w -> cols [w*64,+64)
__global__ __launch_bounds__(256)
void proj_kernel(const _Float16* __restrict__ xs,
                 const _Float16* __restrict__ w16,
                 const float* __restrict__ bias,
                 float* __restrict__ out) {
  const int tid  = threadIdx.x;
  const int wid  = tid >> 5;
  const int lane = tid & 31;
  const int hi   = lane >> 4;
  const int n    = lane & 15;
  const int m0   = blockIdx.x * 32;          // output rows

  v8f acc[2][4];
#pragma unroll
  for (int r = 0; r < 2; ++r)
#pragma unroll
    for (int t = 0; t < 4; ++t) acc[r][t] = v8f{0};

  const _Float16* A0 = xs + ((size_t)(m0 + n)) * CC;
  const _Float16* A1 = xs + ((size_t)(m0 + 16 + n)) * CC;
  for (int kc = 0; kc < CC; kc += 32) {
    HV a0, a1;
    a0.h[0] = *(const v8h*)(A0 + kc + 8 * hi);
    a0.h[1] = *(const v8h*)(A0 + kc + 16 + 8 * hi);
    a1.h[0] = *(const v8h*)(A1 + kc + 8 * hi);
    a1.h[1] = *(const v8h*)(A1 + kc + 16 + 8 * hi);
#pragma unroll
    for (int t = 0; t < 4; ++t) {
      // B[k][n] = W[c0+n][kc+k], row-major contiguous
      const _Float16* bp =
          w16 + ((size_t)(wid * 64 + t * 16 + n)) * CC + kc;
      HV bv;
      bv.h[0] = *(const v8h*)(bp + 8 * hi);
      bv.h[1] = *(const v8h*)(bp + 16 + 8 * hi);
      acc[0][t] = wmma16(a0.v, bv.v, acc[0][t]);
      acc[1][t] = wmma16(a1.v, bv.v, acc[1][t]);
    }
  }
#pragma unroll
  for (int t = 0; t < 4; ++t) {
    int c = wid * 64 + t * 16 + n;
    float bv = bias[c];
#pragma unroll
    for (int r = 0; r < 2; ++r)
#pragma unroll
      for (int j = 0; j < 8; ++j)
        out[((size_t)(m0 + r * 16 + 8 * hi + j)) * CC + c] = acc[r][t][j] + bv;
  }
}

// ---------------------------------------------------------------------------
extern "C" void kernel_launch(void* const* d_in, const int* in_sizes, int n_in,
                              void* d_out, int out_size, void* d_ws, size_t ws_size,
                              hipStream_t stream) {
  const float* dom    = (const float*)d_in[0];
  const float* ref    = (const float*)d_in[1];
  const float* proj_w = (const float*)d_in[2];
  const float* proj_b = (const float*)d_in[3];
  float* out = (float*)d_out;

  const size_t NC = (size_t)BB * NN * CC;          // 8,388,608 elements
  char* ws = (char*)d_ws;
  _Float16* dom16  = (_Float16*)(ws);
  _Float16* ref16  = (_Float16*)(ws + 1 * NC * 2);
  _Float16* refT16 = (_Float16*)(ws + 2 * NC * 2);
  _Float16* xsT16  = (_Float16*)(ws + 3 * NC * 2);
  _Float16* w16    = (_Float16*)(ws + 4 * NC * 2); // + 512*512*2 bytes

  // stage 0: converts
  cvt_f16_kernel<<<1024, 256, 0, stream>>>(dom, dom16, (int)NC);
  cvt_f16_kernel<<<1024, 256, 0, stream>>>(ref, ref16, (int)NC);
  cvt_f16_kernel<<<256, 256, 0, stream>>>(proj_w, w16, CC * CC);
  transpose_cvt_kernel<<<dim3(NN / 32, CC / 32, BB), dim3(32, 8), 0, stream>>>(
      ref, refT16);

  // stage 1: fused attention -> xsT16 (= x^T per batch, == scrambled Xs)
  attn_kernel<<<BB * (NN / 32), 512, 0, stream>>>(dom16, ref16, refT16, xsT16);

  // stage 2: projection
  proj_kernel<<<(BB * NN) / 32, 256, 0, stream>>>(xsT16, w16, proj_b, out);
}
